// LinkPredModel_70806830841994
// MI455X (gfx1250) — matrix operand order repaired and checked by reading, not compiled, not run on hardware
//
#include <hip/hip_runtime.h>
#include <hip/hip_bf16.h>

typedef __attribute__((ext_vector_type(16))) __bf16 v16bf;
typedef __attribute__((ext_vector_type(8)))  float  v8f;

// ---------------------------------------------------------------- utilities

__global__ void zero_f32_kernel(float* __restrict__ p, long n) {
  long i = (long)blockIdx.x * blockDim.x + threadIdx.x;
  if (i < n) p[i] = 0.0f;
}

// Wt[n*64+k] = bf16(W[k*64+n])  (64x64, transpose + convert)
__global__ void prep_wt_kernel(const float* __restrict__ W, __bf16* __restrict__ Wt) {
  int t = blockIdx.x * blockDim.x + threadIdx.x;   // 4096 threads
  int k = t >> 6, n = t & 63;
  Wt[n * 64 + k] = (__bf16)W[k * 64 + n];
}

// ---------------------------------------------------------------- layer 0

__global__ void edge_agg2_kernel(const int* __restrict__ src, const int* __restrict__ dst,
                                 const unsigned char* __restrict__ uf,
                                 const unsigned char* __restrict__ vf,
                                 float* __restrict__ agg0, int E) {
  int e = blockIdx.x * blockDim.x + threadIdx.x;
  if (e >= E) return;
  int s = src[e], d = dst[e];
  if (uf[s]) unsafeAtomicAdd(&agg0[2 * d + 0], 1.0f);
  if (vf[s]) unsafeAtomicAdd(&agg0[2 * d + 1], 1.0f);
}

// h0[n][j] = relu(z0*W0[0][j] + z1*W0[1][j] + b0[j]),  z = (1+eps0)*feat + agg0
__global__ void layer0_kernel(const unsigned char* __restrict__ uf,
                              const unsigned char* __restrict__ vf,
                              const float* __restrict__ agg0,
                              const float* __restrict__ W0, const float* __restrict__ b0,
                              const float* __restrict__ eps0,
                              __bf16* __restrict__ h, int N) {
  long t = (long)blockIdx.x * blockDim.x + threadIdx.x;
  if (t >= (long)N * 64) return;
  int n = (int)(t >> 6), j = (int)(t & 63);
  float e  = 1.0f + eps0[0];
  float z0 = e * (uf[n] ? 1.0f : 0.0f) + agg0[2 * n + 0];
  float z1 = e * (vf[n] ? 1.0f : 0.0f) + agg0[2 * n + 1];
  float y  = z0 * W0[j] + z1 * W0[64 + j] + b0[j];
  h[t] = (__bf16)fmaxf(y, 0.0f);
}

// ---------------------------------------------------------------- GIN aggregation
// h is bf16 (halves the random-gather traffic); agg accumulates in f32 via
// native global_atomic_add_f32. One 8-lane group per edge, b128 (8 bf16)/lane.

__global__ void edge_agg64_kernel(const int* __restrict__ src, const int* __restrict__ dst,
                                  const __bf16* __restrict__ h, float* __restrict__ agg, int E) {
  long t = (long)blockIdx.x * blockDim.x + threadIdx.x;
  int e = (int)(t >> 3);
  if (e >= E) return;
  int c = ((int)t & 7) << 3;                       // 8 elements per lane
  const __bf16* hp = h + (long)src[e] * 64 + c;
  float*        ap = agg + (long)dst[e] * 64 + c;
  uint4 hv = *(const uint4*)hp;                    // 8 bf16 = 16B
  const __bf16* hb = (const __bf16*)&hv;
  #pragma unroll
  for (int i = 0; i < 8; ++i) unsafeAtomicAdd(ap + i, (float)hb[i]);
}

// ---------------------------------------------------------------- WMMA GEMM + bias + ReLU
// Y[M,64] = relu(Z[M,64] @ W[64,64] + bias) as bf16, Wt pre-transposed (Wt[n*64+k]).
// FUSE: Z = (1+eps)*h(bf16) + agg(f32), built in-register (cvt to bf16 A-fragments).
// One wave per 16-row tile: 2 K-steps x 4 col tiles = 8 v_wmma_f32_16x16x32_bf16.
template<bool FUSE>
__global__ __launch_bounds__(256)
void wmma_gemm_relu_kernel(const __bf16* __restrict__ X,
                           const float* __restrict__ agg,
                           const float* __restrict__ eps,
                           const __bf16* __restrict__ Wt,
                           const float* __restrict__ bias,
                           __bf16* __restrict__ Y, int M) {
  int wave = (int)(((long)blockIdx.x * blockDim.x + threadIdx.x) >> 5);
  int lane = threadIdx.x & 31;
  int row0 = wave << 4;
  if (row0 >= M) return;                  // wave-uniform exit (EXEC all-ones at WMMA)
  int ml = lane & 15;                     // A: row M ; B/C/D: col N
  int hi = lane >> 4;                     // half-wave select per ISA 16-bit layouts
  int kA = hi * 8;                        // A lane K-base
  int kB = hi * 8;                        // B lane K-base (same as A: kB+{0..7}, kB+16+{0..7})

  const __bf16* xrow = X + (long)(row0 + ml) * 64;
  const float*  grow = agg + (long)(row0 + ml) * 64;   // unused if !FUSE
  float ec = FUSE ? (1.0f + eps[0]) : 1.0f;

  v8f zero = {0.f, 0.f, 0.f, 0.f, 0.f, 0.f, 0.f, 0.f};
  v8f acc[4] = {zero, zero, zero, zero};

  #pragma unroll
  for (int s = 0; s < 2; ++s) {           // K-steps of 32
    union { uint4 u[2]; __bf16 b[16]; v16bf v; } a;
    #pragma unroll
    for (int ch = 0; ch < 2; ++ch) {      // chunks K = s*32+kA+{0,16} .. +7
      int kk = s * 32 + kA + ch * 16;
      if (FUSE) {
        uint4 hv = *(const uint4*)(xrow + kk);       // 8 bf16
        const __bf16* hb = (const __bf16*)&hv;
        float4 g0 = *(const float4*)(grow + kk);
        float4 g1 = *(const float4*)(grow + kk + 4);
        float gg[8] = {g0.x, g0.y, g0.z, g0.w, g1.x, g1.y, g1.z, g1.w};
        #pragma unroll
        for (int i = 0; i < 8; ++i)
          a.b[ch * 8 + i] = (__bf16)(ec * (float)hb[i] + gg[i]);
      } else {
        a.u[ch] = *(const uint4*)(xrow + kk);
      }
    }
    #pragma unroll
    for (int j = 0; j < 4; ++j) {         // 4 column tiles of 16
      union { uint4 u[2]; v16bf v; } b;
      const __bf16* bp = Wt + (long)(j * 16 + ml) * 64 + s * 32 + kB;
      b.u[0] = *(const uint4*)(bp);
      b.u[1] = *(const uint4*)(bp + 16);
      acc[j] = __builtin_amdgcn_wmma_f32_16x16x32_bf16(
          false, a.v, false, b.v, (short)0, acc[j], false, false);
    }
  }

  // C/D layout: col = ml, row = vgpr + 8*hi
  #pragma unroll
  for (int j = 0; j < 4; ++j) {
    int col = j * 16 + ml;
    float bv = bias[col];
    #pragma unroll
    for (int v = 0; v < 8; ++v) {
      int row = row0 + v + hi * 8;
      Y[(long)row * 64 + col] = (__bf16)fmaxf(acc[j][v] + bv, 0.0f);
    }
  }
}

// ---------------------------------------------------------------- pooling + head

__global__ void pool_accum_kernel(const __bf16* __restrict__ h, const int* __restrict__ n2g,
                                  float* __restrict__ sums, float* __restrict__ cnts, int N) {
  long t = (long)blockIdx.x * blockDim.x + threadIdx.x;
  if (t >= (long)N * 64) return;
  int n = (int)(t >> 6), j = (int)(t & 63);
  int g = n2g[n];
  unsafeAtomicAdd(&sums[(long)g * 64 + j], (float)h[t]);
  if (j == 0) unsafeAtomicAdd(&cnts[g], 1.0f);
}

__global__ void pool_div_kernel(const float* __restrict__ sums, const float* __restrict__ cnts,
                                __bf16* __restrict__ hgb, int Bn) {
  int t = blockIdx.x * blockDim.x + threadIdx.x;
  if (t >= Bn * 64) return;
  int g = t >> 6;
  hgb[t] = (__bf16)(sums[t] / fmaxf(cnts[g], 1.0f));
}

__global__ void score_kernel(const __bf16* __restrict__ s1, const float* __restrict__ Ws2,
                             const float* __restrict__ bs2, float* __restrict__ out, int Bn) {
  int g = blockIdx.x * blockDim.x + threadIdx.x;
  if (g >= Bn) return;
  float acc = bs2[0];
  #pragma unroll 8
  for (int j = 0; j < 64; ++j) acc += (float)s1[g * 64 + j] * Ws2[j];
  out[g] = 1.0f / (1.0f + __expf(-acc));
}

// ---------------------------------------------------------------- driver

extern "C" void kernel_launch(void* const* d_in, const int* in_sizes, int n_in,
                              void* d_out, int out_size, void* d_ws, size_t ws_size,
                              hipStream_t stream) {
  const unsigned char* uf  = (const unsigned char*)d_in[0];  // bool
  const unsigned char* vf  = (const unsigned char*)d_in[1];  // bool
  const int*   src  = (const int*)d_in[2];
  const int*   dst  = (const int*)d_in[3];
  const int*   n2g  = (const int*)d_in[4];
  const float* W0   = (const float*)d_in[5];
  const float* b0   = (const float*)d_in[6];
  const float* eps0 = (const float*)d_in[7];
  const float* W1   = (const float*)d_in[8];
  const float* b1   = (const float*)d_in[9];
  const float* eps1 = (const float*)d_in[10];
  const float* W2   = (const float*)d_in[11];
  const float* b2   = (const float*)d_in[12];
  const float* eps2 = (const float*)d_in[13];
  const float* Ws1  = (const float*)d_in[14];
  const float* bs1  = (const float*)d_in[15];
  const float* Ws2  = (const float*)d_in[16];
  const float* bs2  = (const float*)d_in[17];

  const int N  = in_sizes[0];
  const int E  = in_sizes[2];
  const int Bn = out_size;           // 1024 graphs
  float* out   = (float*)d_out;

  // workspace carve-out (256B aligned)
  char* ws = (char*)d_ws;
  auto carve = [&](size_t bytes) { char* p = ws; ws += (bytes + 255) & ~(size_t)255; return p; };
  __bf16* h    = (__bf16*)carve((size_t)N * 64 * 2);
  float*  agg  = (float*) carve((size_t)N * 64 * 4);
  float*  agg0 = (float*) carve((size_t)N * 2 * 4);
  __bf16* W1t  = (__bf16*)carve(4096 * 2);
  __bf16* W2t  = (__bf16*)carve(4096 * 2);
  __bf16* Ws1t = (__bf16*)carve(4096 * 2);
  float*  sums = (float*) carve((size_t)Bn * 64 * 4);
  float*  cnts = (float*) carve((size_t)Bn * 4);
  __bf16* hgb  = (__bf16*)carve((size_t)Bn * 64 * 2);
  __bf16* s1   = (__bf16*)carve((size_t)Bn * 64 * 2);
  (void)ws_size; (void)n_in;

  const int TB = 256;
  const long NH = (long)N * 64;
  dim3 gNH((unsigned)((NH + TB - 1) / TB));
  dim3 gE8((unsigned)(((long)E * 8 + TB - 1) / TB));
  dim3 gGemmN((unsigned)((N + 127) / 128));   // 8 waves/block, 16 rows/wave

  // weight prep (transpose + bf16)
  prep_wt_kernel<<<16, TB, 0, stream>>>(W1,  W1t);
  prep_wt_kernel<<<16, TB, 0, stream>>>(W2,  W2t);
  prep_wt_kernel<<<16, TB, 0, stream>>>(Ws1, Ws1t);

  // ---- layer 0 (K=2, scalar path)
  zero_f32_kernel<<<(unsigned)(((long)N * 2 + TB - 1) / TB), TB, 0, stream>>>(agg0, (long)N * 2);
  edge_agg2_kernel<<<(unsigned)((E + TB - 1) / TB), TB, 0, stream>>>(src, dst, uf, vf, agg0, E);
  layer0_kernel<<<gNH, TB, 0, stream>>>(uf, vf, agg0, W0, b0, eps0, h, N);

  // ---- layers 1 & 2 (WMMA path, fused z-construction)
  const float* Weps[2] = {eps1, eps2};
  const float* Wb[2]   = {b1, b2};
  __bf16*      Wt[2]   = {W1t, W2t};
  for (int l = 0; l < 2; ++l) {
    zero_f32_kernel<<<gNH, TB, 0, stream>>>(agg, NH);
    edge_agg64_kernel<<<gE8, TB, 0, stream>>>(src, dst, h, agg, E);
    wmma_gemm_relu_kernel<true><<<gGemmN, TB, 0, stream>>>(h, agg, Weps[l], Wt[l], Wb[l], h, N);
  }

  // ---- mean pooling
  zero_f32_kernel<<<(unsigned)(((long)Bn * 64 + TB - 1) / TB), TB, 0, stream>>>(sums, (long)Bn * 64);
  zero_f32_kernel<<<(unsigned)((Bn + TB - 1) / TB), TB, 0, stream>>>(cnts, Bn);
  pool_accum_kernel<<<gNH, TB, 0, stream>>>(h, n2g, sums, cnts, N);
  pool_div_kernel<<<(unsigned)((Bn * 64 + TB - 1) / TB), TB, 0, stream>>>(sums, cnts, hgb, Bn);

  // ---- scorer: WMMA GEMM (1024x64x64) + relu, then dot + sigmoid
  wmma_gemm_relu_kernel<false><<<(unsigned)((Bn + 127) / 128), TB, 0, stream>>>(
      hgb, nullptr, nullptr, Ws1t, bs1, s1, Bn);
  score_kernel<<<(unsigned)((Bn + TB - 1) / TB), TB, 0, stream>>>(s1, Ws2, bs2, out, Bn);
}
